// entropy_45810121179229
// MI455X (gfx1250) — matrix operand order repaired and checked by reading, not compiled, not run on hardware
//
#include <hip/hip_runtime.h>

typedef __attribute__((ext_vector_type(16))) _Float16 v16h;
typedef __attribute__((ext_vector_type(8)))  float    v8f;

#define NROWS 8192
#define KDIM  768
#define BM    128
#define BN    256
#define LDK   40      // padded LDS row stride (halves): 80B, 16B-aligned chunks
#define KSTEPS (KDIM / 32)

// ---- order-preserving float <-> uint32 encoding (for integer atomicMax) ----
__device__ __forceinline__ unsigned encodeF(float f) {
    unsigned u = __float_as_uint(f);
    return (u & 0x80000000u) ? ~u : (u | 0x80000000u);
}
__device__ __forceinline__ float decodeF(unsigned e) {
    unsigned u = (e & 0x80000000u) ? (e & 0x7FFFFFFFu) : ~e;
    return __uint_as_float(u);
}

// ---- CDNA5 async global->LDS copy (ASYNCcnt-tracked, no VGPR round-trip) ----
// LDS generic pointers on AMDGPU carry the 0-based LDS offset in the low 32 bits.
__device__ __forceinline__ unsigned ldsAddrOf(const void* p) {
    return (unsigned)(size_t)p;
}
__device__ __forceinline__ void async_cp16(unsigned lds_addr, const _Float16* g) {
    asm volatile("global_load_async_to_lds_b128 %0, %1, off"
                 :: "v"(lds_addr), "v"(g)
                 : "memory");
}

// ---------------------------------------------------------------------------
// 1) Row-normalize f32 -> f16 (cosine prep), one block (256 thr) per row.
// ---------------------------------------------------------------------------
__global__ __launch_bounds__(256) void normalize_kernel(const float* __restrict__ src,
                                                        _Float16* __restrict__ dst) {
    const int row = blockIdx.x;
    const int tid = threadIdx.x;
    const float* p = src + (size_t)row * KDIM;

    float v0 = p[tid];
    float v1 = p[tid + 256];
    float v2 = p[tid + 512];
    float s  = v0 * v0 + v1 * v1 + v2 * v2;

    __shared__ float red[256];
    red[tid] = s;
    __syncthreads();
    for (int o = 128; o > 0; o >>= 1) {
        if (tid < o) red[tid] += red[tid + o];
        __syncthreads();
    }
    const float inv = 1.0f / fmaxf(sqrtf(red[0]), 1e-8f);

    _Float16* q = dst + (size_t)row * KDIM;
    q[tid]       = (_Float16)(v0 * inv);
    q[tid + 256] = (_Float16)(v1 * inv);
    q[tid + 512] = (_Float16)(v2 * inv);
}

// ---------------------------------------------------------------------------
// 2) Zero the encoded-max accumulators (16384 u32: rowKeys then colKeys).
// ---------------------------------------------------------------------------
__global__ void init_keys_kernel(unsigned* __restrict__ keys) {
    keys[blockIdx.x * 256 + threadIdx.x] = 0u;
}

// ---------------------------------------------------------------------------
// 3) Fused GEMM + row/col max.
//    Block = 256 threads = 8 waves; block tile 128x256; wave tile 64x64
//    (4x4 accumulators of V_WMMA_F32_16X16X32_F16).
//    Double-buffered LDS fed by global_load_async_to_lds_b128.
// ---------------------------------------------------------------------------
__global__ __launch_bounds__(256) void gemm_max_kernel(const _Float16* __restrict__ xh,
                                                       const _Float16* __restrict__ yh,
                                                       unsigned* __restrict__ rowKeys,
                                                       unsigned* __restrict__ colKeys) {
    __shared__ __align__(16) _Float16 As[2][BM * LDK];
    __shared__ __align__(16) _Float16 Bs[2][BN * LDK];
    __shared__ unsigned rowLds[BM];
    __shared__ unsigned colLds[BN];

    const int tid  = threadIdx.x;
    const int lane = tid & 31;
    const int wave = tid >> 5;
    const int lm   = lane & 15;   // row/col within 16x16 tile
    const int h    = lane >> 4;   // lane half-group

    const int waveRow = (wave & 1) * 64;    // 0 or 64
    const int waveCol = (wave >> 1) * 64;   // 0,64,128,192
    const size_t rowBase = (size_t)blockIdx.y * BM;
    const size_t colBase = (size_t)blockIdx.x * BN;

    if (tid < BM) rowLds[tid] = 0u;
    colLds[tid] = 0u;             // BN == blockDim.x == 256

    v8f acc[4][4];
    const v8f vzero = {0.f, 0.f, 0.f, 0.f, 0.f, 0.f, 0.f, 0.f};
#pragma unroll
    for (int mi = 0; mi < 4; ++mi)
#pragma unroll
        for (int ni = 0; ni < 4; ++ni) acc[mi][ni] = vzero;

    // Stage one 32-wide k-slab of both tiles into LDS buffer `buf`
    // (6 async b128 copies per thread: 2 for A rows 0..127, 4 for B rows 0..255).
    auto stage = [&](int k0, int buf) {
#pragma unroll
        for (int j = 0; j < 2; ++j) {
            const int c = tid + j * 256;
            const int r = c >> 2, q = (c & 3) * 8;
            async_cp16(ldsAddrOf(&As[buf][r * LDK + q]),
                       &xh[(rowBase + r) * KDIM + k0 + q]);
        }
#pragma unroll
        for (int j = 0; j < 4; ++j) {
            const int c = tid + j * 256;
            const int r = c >> 2, q = (c & 3) * 8;
            async_cp16(ldsAddrOf(&Bs[buf][r * LDK + q]),
                       &yh[(colBase + r) * KDIM + k0 + q]);
        }
    };

    stage(0, 0);

    for (int it = 0; it < KSTEPS; ++it) {
        const int k0  = it * 32;
        const int buf = it & 1;

        if (it + 1 < KSTEPS) {
            stage(k0 + 32, buf ^ 1);                 // overlap copy of tile it+1
            if (k0 + 64 < KDIM) {                    // L2 warm-up for tile it+2
                __builtin_prefetch(&xh[(rowBase + (tid >> 2)) * KDIM + k0 + 64], 0, 1);
                __builtin_prefetch(&yh[(colBase + (tid >> 2)) * KDIM + k0 + 64], 0, 1);
            }
            // 6 new asyncs in flight; retire the 6 belonging to tile `it`
            asm volatile("s_wait_asynccnt 0x6" ::: "memory");
        } else {
            asm volatile("s_wait_asynccnt 0x0" ::: "memory");
        }
        __syncthreads();   // tile `it` visible to all waves

        // ---- B fragments (32x16 f16): lane holds col N=lm,
        //      halves 0..15 = K[16h .. 16h+15] (contiguous 32B) ----
        v16h bfrag[4];
#pragma unroll
        for (int ni = 0; ni < 4; ++ni) {
            const _Float16* pb = &Bs[buf][(waveCol + ni * 16 + lm) * LDK + 16 * h];
            union { uint4 u[2]; v16h v; } t;
            t.u[0] = *(const uint4*)pb;
            t.u[1] = *(const uint4*)(pb + 8);
            bfrag[ni] = t.v;
        }
        // ---- A fragments (16x32 f16): lane holds row M=lm,
        //      halves 0..15 = K[16h .. 16h+15] (contiguous 32B) ----
#pragma unroll
        for (int mi = 0; mi < 4; ++mi) {
            const _Float16* pa = &As[buf][(waveRow + mi * 16 + lm) * LDK + 16 * h];
            union { uint4 u[2]; v16h v; } t;
            t.u[0] = *(const uint4*)pa;
            t.u[1] = *(const uint4*)(pa + 8);
            const v16h afrag = t.v;
#pragma unroll
            for (int ni = 0; ni < 4; ++ni)
                acc[mi][ni] = __builtin_amdgcn_wmma_f32_16x16x32_f16(
                    false, afrag, false, bfrag[ni],
                    (short)0, acc[mi][ni], false, false);
        }
        __syncthreads();   // all waves done reading buf before it is re-staged
    }

    // ---- fold maxima: pre-reduce in registers, then ds_max_u32 ----
    // C/D layout: lane (g=h, n=lm), VGPR r -> element (M = r + 8h, N = n).
#pragma unroll
    for (int mi = 0; mi < 4; ++mi)
#pragma unroll
        for (int r = 0; r < 8; ++r) {
            float v = acc[mi][0][r];
#pragma unroll
            for (int ni = 1; ni < 4; ++ni) v = fmaxf(v, acc[mi][ni][r]);
            atomicMax(&rowLds[waveRow + mi * 16 + r + 8 * h], encodeF(v));
        }
#pragma unroll
    for (int ni = 0; ni < 4; ++ni) {
        float v = acc[0][ni][0];
#pragma unroll
        for (int mi = 0; mi < 4; ++mi)
#pragma unroll
            for (int r = 0; r < 8; ++r) v = fmaxf(v, acc[mi][ni][r]);
        atomicMax(&colLds[waveCol + ni * 16 + lm], encodeF(v));
    }
    __syncthreads();

    // ---- one global atomic per row/col per block ----
    if (tid < BM) atomicMax(&rowKeys[rowBase + tid], rowLds[tid]);
    atomicMax(&colKeys[colBase + tid], colLds[tid]);
}

// ---------------------------------------------------------------------------
// 4) Entropy reduction: block 0 -> out[0] (row maxima), block 1 -> out[1].
// ---------------------------------------------------------------------------
__global__ __launch_bounds__(256) void finalize_kernel(const unsigned* __restrict__ keys,
                                                       float* __restrict__ out) {
    const unsigned* k = keys + (size_t)blockIdx.x * NROWS;
    const int tid = threadIdx.x;
    float s = 0.0f;
    for (int i = tid; i < NROWS; i += 256) {
        const float c = decodeF(k[i]);
        float p = 1.0f - erff((1.0f - c) * 0.70710678118654752f);
        p = fmaxf(p, 1e-30f);
        s += -(log2f(p) * p);
    }
    __shared__ float red[256];
    red[tid] = s;
    __syncthreads();
    for (int o = 128; o > 0; o >>= 1) {
        if (tid < o) red[tid] += red[tid + o];
        __syncthreads();
    }
    if (tid == 0) out[blockIdx.x] = red[0];
}

// ---------------------------------------------------------------------------
extern "C" void kernel_launch(void* const* d_in, const int* in_sizes, int n_in,
                              void* d_out, int out_size, void* d_ws, size_t ws_size,
                              hipStream_t stream) {
    const float* ex = (const float*)d_in[0];
    const float* ey = (const float*)d_in[1];
    float* out = (float*)d_out;

    // workspace: xh[8192*768 f16] | yh[8192*768 f16] | rowKeys[8192] | colKeys[8192]
    _Float16* xh = (_Float16*)d_ws;
    _Float16* yh = xh + (size_t)NROWS * KDIM;
    unsigned* rowKeys = (unsigned*)(yh + (size_t)NROWS * KDIM);
    unsigned* colKeys = rowKeys + NROWS;

    normalize_kernel<<<NROWS, 256, 0, stream>>>(ex, xh);
    normalize_kernel<<<NROWS, 256, 0, stream>>>(ey, yh);
    init_keys_kernel<<<(2 * NROWS) / 256, 256, 0, stream>>>(rowKeys);

    dim3 grid(NROWS / BN, NROWS / BM);   // (32, 64)
    gemm_max_kernel<<<grid, 256, 0, stream>>>(xh, yh, rowKeys, colKeys);

    finalize_kernel<<<2, 256, 0, stream>>>(rowKeys, out);
}
